// Money_former_MLA_11312943858151
// MI455X (gfx1250) — matrix operand order repaired and checked by reading, not compiled, not run on hardware
//
#include <hip/hip_runtime.h>
#include <hip/hip_bf16.h>

// ---------------------------------------------------------------------------
// Types for CDNA5 WMMA (wave32) + TDM descriptors
// ---------------------------------------------------------------------------
typedef __bf16 bf16_t;
typedef __attribute__((ext_vector_type(16))) __bf16 v16bf;
typedef __attribute__((ext_vector_type(8)))  __bf16 v8bf;
typedef __attribute__((ext_vector_type(8)))  float  v8f;
typedef unsigned int u32x4 __attribute__((ext_vector_type(4)));
typedef int          i32x8 __attribute__((ext_vector_type(8)));
typedef int          i32x4 __attribute__((ext_vector_type(4)));

// ---------------------------------------------------------------------------
// Model constants
// ---------------------------------------------------------------------------
static constexpr int NTICK   = 8;
static constexpr int NFEAT   = 64;
static constexpr int DMODEL  = 1024;
static constexpr int NHEAD   = 16;
static constexpr int HD      = 128;
static constexpr int KVC     = 512;
static constexpr int QC      = 768;
static constexpr int DFF     = 2688;
static constexpr int SPS     = 256;            // SEQ + 1
static constexpr int LTOT    = SPS * NTICK;    // 2048
static constexpr float EPS_DEF  = 1.1920929e-07f;
static constexpr float EPS_ATTN = 1e-05f;
static constexpr float ATTN_SCALE = 0.07216878364870322f; // (HD+RD)^-0.5

// ---------------------------------------------------------------------------
// f32 -> bf16 conversion (weights, once per launch)
// ---------------------------------------------------------------------------
__global__ void k_f32_to_bf16(const float* __restrict__ in, bf16_t* __restrict__ out, int n) {
    int i = blockIdx.x * 256 + threadIdx.x;
    if (i < n) out[i] = (bf16_t)in[i];
}

// ---------------------------------------------------------------------------
// Embedding: h[l,d] built from sep/x/shared/unique/ticker, scaled by 1/sqrt(D)
// ---------------------------------------------------------------------------
__global__ void k_embed(const float* __restrict__ x,
                        const int*   __restrict__ seperator,
                        const int*   __restrict__ tickers,
                        const float* __restrict__ shared_w,
                        const float* __restrict__ shared_b,
                        const float* __restrict__ unique_w,
                        const float* __restrict__ unique_b,
                        const float* __restrict__ sep_emb,
                        const float* __restrict__ ticker_emb,
                        float* __restrict__ h) {
    int idx = blockIdx.x * 256 + threadIdx.x;
    if (idx >= LTOT * DMODEL) return;
    int l = idx >> 10;
    int d = idx & (DMODEL - 1);
    int s = l >> 3;
    int t = l & 7;
    float base;
    if (s == 0) {
        base = sep_emb[seperator[0] * DMODEL + d];
    } else {
        const float* xr = x + ((size_t)(s - 1) * NTICK + t) * NFEAT;
        float acc;
        if (d < 768) {
            const float* w = shared_w + d;
            acc = shared_b[d];
            #pragma unroll 8
            for (int f = 0; f < NFEAT; ++f) acc += xr[f] * w[(size_t)f * 768];
        } else {
            int du = d - 768;
            const float* w = unique_w + (size_t)t * NFEAT * 256 + du;
            acc = unique_b[t * 256 + du];
            #pragma unroll 8
            for (int f = 0; f < NFEAT; ++f) acc += xr[f] * w[(size_t)f * 256];
        }
        base = acc;
    }
    h[idx] = (base + ticker_emb[tickers[t] * DMODEL + d]) * 0.03125f; // 1/sqrt(1024)
}

// ---------------------------------------------------------------------------
// RMSNorm: f32 rows -> bf16 rows (packed), one block (128 thr) per row
// ---------------------------------------------------------------------------
__global__ void k_rmsnorm_bf16(const float* __restrict__ in, int instride,
                               bf16_t* __restrict__ out,
                               const float* __restrict__ w, int cols, float eps) {
    int row = blockIdx.x;
    int t = threadIdx.x;
    __shared__ float red[128];
    const float* r = in + (size_t)row * instride;
    float ss = 0.f;
    for (int c = t; c < cols; c += 128) { float v = r[c]; ss += v * v; }
    red[t] = ss; __syncthreads();
    for (int s = 64; s > 0; s >>= 1) { if (t < s) red[t] += red[t + s]; __syncthreads(); }
    float scale = rsqrtf(red[0] / (float)cols + eps);
    for (int c = t; c < cols; c += 128)
        out[(size_t)row * cols + c] = (bf16_t)(r[c] * scale * w[c]);
}

// ---------------------------------------------------------------------------
// TDM: issue a 2D tensor_load_to_lds of a [rows x 32] bf16 tile from a
// row-major tensor with element stride `lda` into contiguous LDS.
// Descriptor layout per CDNA5 ISA 8.3/8.4 (group0 128b, group1 256b).
// This toolchain's builtin takes 6 args (g0,g1,g2,g3,extra,cpol).
// ---------------------------------------------------------------------------
__device__ __forceinline__ void tdm_load_tile(const bf16_t* gptr, unsigned lds_addr,
                                              int lda, unsigned tile_rows) {
    unsigned long long ga = (unsigned long long)(uintptr_t)gptr;
    u32x4 g0;
    g0[0] = 1u;                                              // count=1 (valid user D#)
    g0[1] = lds_addr;                                        // LDS byte address
    g0[2] = (unsigned)ga;                                    // global_addr[31:0]
    g0[3] = (unsigned)((ga >> 32) & 0x01FFFFFFu) | 0x80000000u; // addr[56:32] | type=2
    unsigned dim0 = (unsigned)lda;                           // tensor dim0 (elements)
    unsigned dim1 = 1u << 20;                                // tensor dim1 (large, no OOB)
    unsigned long long s0 = (unsigned long long)(unsigned)lda; // dim0 stride (elements)
    i32x8 g1;
    g1[0] = (int)(1u << 16);                                 // data_size=1 -> 2 bytes
    g1[1] = (int)((dim0 & 0xFFFFu) << 16);                   // tensor_dim0[15:0]
    g1[2] = (int)((dim0 >> 16) | ((dim1 & 0xFFFFu) << 16));  // dim0[31:16] | dim1[15:0]
    g1[3] = (int)((dim1 >> 16) | (32u << 16));               // dim1[31:16] | tile_dim0=32
    g1[4] = (int)(tile_rows & 0xFFFFu);                      // tile_dim1 | tile_dim2=0
    g1[5] = (int)(unsigned)(s0 & 0xFFFFFFFFull);             // dim0_stride[31:0]
    g1[6] = (int)(unsigned)(s0 >> 32);                       // dim0_stride[47:32] | s1 lo
    g1[7] = 0;                                               // dim1_stride hi (unused, 2D)
    i32x4 z4 = {0, 0, 0, 0};
    i32x8 z8 = {0, 0, 0, 0, 0, 0, 0, 0};
    __builtin_amdgcn_tensor_load_to_lds(g0, g1, z4, z4, z8, 0);
}

// ---------------------------------------------------------------------------
// Generic tiled WMMA GEMM:  C[M,N] = alpha * A[M,K] x B + bias
//   A: bf16 row-major (lda).  B: bf16, row-major KxN (TRANSB=false)
//   or row-major NxK (TRANSB=true, computes A.B^T).  C: f32 (ldc).
//   Requires M%128==0, N%64==0, K%32==0 (all shapes here satisfy this).
//   A-tile is staged global->LDS by the Tensor Data Mover (wave 0 issues the
//   D#, waits on TENSORcnt, barrier releases consumers). B-tile is staged by
//   vector loads into a padded transposed LDS tile.
//   Block: 256 threads = 8 waves; wave (wm,wn) owns a 32x32 tile = 2x2 frags.
// ---------------------------------------------------------------------------
static constexpr int BM = 128, BN = 64, BK = 32;
static constexpr int SARow = 32;   // A LDS row stride (64B, 16B-aligned frag loads)
static constexpr int SBRow = 40;   // B LDS row stride (80B, padded, 16B-aligned)

template<bool TRANSB>
__global__ __launch_bounds__(256) void k_gemm_wmma(
    const bf16_t* __restrict__ A, int lda,
    const bf16_t* __restrict__ B, int ldb,
    float* __restrict__ C, int ldc,
    const float* __restrict__ bias,
    int M, int N, int K, float alpha) {
    __shared__ __attribute__((aligned(16))) bf16_t sA[BM * SARow];
    __shared__ __attribute__((aligned(16))) bf16_t sB[BN * SBRow];
    const int tid  = threadIdx.x;
    const int wave = tid >> 5, lane = tid & 31;
    const int wm = wave & 3, wn = wave >> 2;
    const int m0 = blockIdx.y * BM, n0 = blockIdx.x * BN;
    const int half = lane >> 4, l16 = lane & 15;
    const unsigned ldsA_addr = (unsigned)(uintptr_t)(&sA[0]); // addr[31:0] = LDS offset

    v8f acc[2][2] = {};

    for (int k0 = 0; k0 < K; k0 += BK) {
        // ---- stage A tile (128x32) via Tensor Data Mover ----
        if (wave == 0) {
            tdm_load_tile(A + (size_t)m0 * lda + k0, ldsA_addr, lda, BM);
        }
        // ---- stage B tile into sB[n][k] (transposed-in-LDS) ----
        if (TRANSB) {
            int n = tid >> 2, seg = (tid & 3) * 8;
            const uint2* g = reinterpret_cast<const uint2*>(
                B + (size_t)(n0 + n) * ldb + k0 + seg);
            uint2* s = reinterpret_cast<uint2*>(&sB[n * SBRow + seg]);
            s[0] = g[0]; s[1] = g[1];
        } else {
            int k = tid >> 3, nseg = (tid & 7) * 8;
            const bf16_t* g = B + (size_t)(k0 + k) * ldb + n0 + nseg;
            bf16_t tmp[8];
            *reinterpret_cast<uint2*>(&tmp[0]) = *reinterpret_cast<const uint2*>(g);
            *reinterpret_cast<uint2*>(&tmp[4]) = *reinterpret_cast<const uint2*>(g + 4);
            #pragma unroll
            for (int e = 0; e < 8; ++e) sB[(nseg + e) * SBRow + k] = tmp[e];
        }
        if (k0 + BK < K)
            __builtin_prefetch(B, 0, 1);
        // wave 0: DMA must have landed before anyone reads sA
        if (wave == 0)
            __builtin_amdgcn_s_wait_tensorcnt(0);
        __syncthreads();

        // ---- load fragments per ISA VGPR layouts ----
        v16bf afrag[2], bfrag[2];
        #pragma unroll
        for (int i = 0; i < 2; ++i) {
            int row = wm * 32 + i * 16 + l16;
            int kb = half * 8;   // lanes0-15: K {0..7,16..23}; lanes16-31: {8..15,24..31}
            const v8bf lo = *reinterpret_cast<const v8bf*>(&sA[row * SARow + kb]);
            const v8bf hi = *reinterpret_cast<const v8bf*>(&sA[row * SARow + kb + 16]);
            v16bf a;
            #pragma unroll
            for (int e = 0; e < 8; ++e) { a[e] = lo[e]; a[e + 8] = hi[e]; }
            afrag[i] = a;
        }
        #pragma unroll
        for (int j = 0; j < 2; ++j) {
            int col = wn * 32 + j * 16 + l16;
            int kb = half * 8;   // lanes0-15: K {0..7,16..23}; lanes16-31: {8..15,24..31}
            const v8bf lo = *reinterpret_cast<const v8bf*>(&sB[col * SBRow + kb]);
            const v8bf hi = *reinterpret_cast<const v8bf*>(&sB[col * SBRow + kb + 16]);
            v16bf b;
            #pragma unroll
            for (int e = 0; e < 8; ++e) { b[e] = lo[e]; b[e + 8] = hi[e]; }
            bfrag[j] = b;
        }
        #pragma unroll
        for (int i = 0; i < 2; ++i)
            #pragma unroll
            for (int j = 0; j < 2; ++j)
                acc[i][j] = __builtin_amdgcn_wmma_f32_16x16x32_bf16(
                    false, afrag[i], false, bfrag[j], (short)0, acc[i][j], false, false);
        __syncthreads();   // all reads of sA/sB done before next iteration's DMA/stores
    }

    // ---- epilogue: C layout VGPR r -> row r + 8*half, col = lane&15 ----
    #pragma unroll
    for (int i = 0; i < 2; ++i) {
        #pragma unroll
        for (int j = 0; j < 2; ++j) {
            int col = n0 + wn * 32 + j * 16 + l16;
            float bv = bias ? bias[col] : 0.0f;
            #pragma unroll
            for (int r = 0; r < 8; ++r) {
                int row = m0 + wm * 32 + i * 16 + r + half * 8;
                C[(size_t)row * ldc + col] = acc[i][j][r] * alpha + bv;
            }
        }
    }
}

// ---------------------------------------------------------------------------
// Build K_attn (bf16, [head][l][192]): first 128 from kvup, last 64 = RoPE(k_rope)
// ---------------------------------------------------------------------------
__global__ void k_build_k(const float* __restrict__ kv, const float* __restrict__ ckv,
                          const float* __restrict__ cosb, const float* __restrict__ sinb,
                          bf16_t* __restrict__ Kattn) {
    int idx = blockIdx.x * 256 + threadIdx.x;
    if (idx >= NHEAD * LTOT * 192) return;
    int h = idx / (LTOT * 192);
    int rem = idx - h * (LTOT * 192);
    int l = rem / 192;
    int d = rem - l * 192;
    float val;
    if (d < HD) {
        val = kv[(size_t)l * 4096 + h * 256 + d];
    } else {
        int j = d - HD;
        int s = l >> 3;
        const float* kr = ckv + (size_t)l * 576 + KVC;
        if (s == 0) {
            val = kr[j];
        } else {
            int i = j >> 1;
            float c  = cosb[(s - 1) * 32 + i];
            float sn = sinb[(s - 1) * 32 + i];
            float xr = kr[2 * i], xi = kr[2 * i + 1];
            val = (j & 1) ? (xr * sn + xi * c) : (xr * c - xi * sn);
        }
    }
    Kattn[idx] = (bf16_t)val;
}

__global__ void k_build_v(const float* __restrict__ kv, bf16_t* __restrict__ Vattn) {
    int idx = blockIdx.x * 256 + threadIdx.x;
    if (idx >= NHEAD * LTOT * HD) return;
    int h = idx / (LTOT * HD);
    int rem = idx - h * (LTOT * HD);
    int l = rem / HD;
    int d = rem - l * HD;
    Vattn[idx] = (bf16_t)kv[(size_t)l * 4096 + h * 256 + HD + d];
}

__global__ void k_build_q(const float* __restrict__ qf,
                          const float* __restrict__ cosb, const float* __restrict__ sinb,
                          bf16_t* __restrict__ Qattn) {
    int idx = blockIdx.x * 256 + threadIdx.x;
    if (idx >= NHEAD * LTOT * 192) return;
    int h = idx / (LTOT * 192);
    int rem = idx - h * (LTOT * 192);
    int l = rem / 192;
    int d = rem - l * 192;
    const float* qr = qf + (size_t)l * 3072 + h * 192;
    float val;
    if (d < HD) {
        val = qr[d];
    } else {
        int j = d - HD;
        int s = l >> 3;
        if (s == 0) {
            val = qr[HD + j];
        } else {
            int i = j >> 1;
            float c  = cosb[(s - 1) * 32 + i];
            float sn = sinb[(s - 1) * 32 + i];
            float xr = qr[HD + 2 * i], xi = qr[HD + 2 * i + 1];
            val = (j & 1) ? (xr * sn + xi * c) : (xr * c - xi * sn);
        }
    }
    Qattn[idx] = (bf16_t)val;
}

// ---------------------------------------------------------------------------
// Masked softmax (tiled-causal mask: disallow (k%256) > (q%256)), bf16 probs
// ---------------------------------------------------------------------------
__global__ void k_softmax(const float* __restrict__ S, bf16_t* __restrict__ P, float scale) {
    int q = blockIdx.x;
    int t = threadIdx.x;               // 256 threads, 2048 cols
    __shared__ float red[256];
    int qm = q & 255;
    float vals[8];
    float mx = -3.0e38f;
    #pragma unroll
    for (int i = 0; i < 8; ++i) {
        int c = t + i * 256;
        float v = S[(size_t)q * LTOT + c] * scale;
        if ((c & 255) > qm) v = -1.0e9f;
        vals[i] = v;
        mx = fmaxf(mx, v);
    }
    red[t] = mx; __syncthreads();
    for (int s = 128; s > 0; s >>= 1) { if (t < s) red[t] = fmaxf(red[t], red[t + s]); __syncthreads(); }
    mx = red[0]; __syncthreads();
    float sum = 0.f;
    #pragma unroll
    for (int i = 0; i < 8; ++i) { vals[i] = __expf(vals[i] - mx); sum += vals[i]; }
    red[t] = sum; __syncthreads();
    for (int s = 128; s > 0; s >>= 1) { if (t < s) red[t] += red[t + s]; __syncthreads(); }
    float inv = 1.f / red[0];
    #pragma unroll
    for (int i = 0; i < 8; ++i) {
        int c = t + i * 256;
        P[(size_t)q * LTOT + c] = (bf16_t)(vals[i] * inv);
    }
}

// ---------------------------------------------------------------------------
// SwiGLU gate: gated = bf16( u * silu(v) ), u=g[:,:DFF], v=g[:,DFF:]
// ---------------------------------------------------------------------------
__global__ void k_swiglu(const float* __restrict__ g, bf16_t* __restrict__ gated) {
    int idx = blockIdx.x * 256 + threadIdx.x;
    if (idx >= LTOT * DFF) return;
    int l = idx / DFF;
    int j = idx - l * DFF;
    float u = g[(size_t)l * (2 * DFF) + j];
    float v = g[(size_t)l * (2 * DFF) + DFF + j];
    float sv = v / (1.f + __expf(-v));
    gated[idx] = (bf16_t)(u * sv);
}

// ---------------------------------------------------------------------------
// Residual add
// ---------------------------------------------------------------------------
__global__ void k_add(float* __restrict__ h, const float* __restrict__ a, int n) {
    int i = blockIdx.x * 256 + threadIdx.x;
    if (i < n) h[i] += a[i];
}

// ---------------------------------------------------------------------------
// Final rmsnorm + out matvec (N=8 too small for WMMA tiles)
// ---------------------------------------------------------------------------
__global__ void k_final_out(const float* __restrict__ h, const float* __restrict__ w,
                            const float* __restrict__ out_w, const float* __restrict__ out_b,
                            float* __restrict__ out) {
    int l = blockIdx.x;
    int t = threadIdx.x;     // 128 threads
    __shared__ float red[128];
    const float* hr = h + (size_t)l * DMODEL;
    float ss = 0.f;
    for (int c = t; c < DMODEL; c += 128) { float v = hr[c]; ss += v * v; }
    red[t] = ss; __syncthreads();
    for (int s = 64; s > 0; s >>= 1) { if (t < s) red[t] += red[t + s]; __syncthreads(); }
    float scale = rsqrtf(red[0] / (float)DMODEL + EPS_DEF);
    __syncthreads();
    for (int j = 0; j < 8; ++j) {
        float acc = 0.f;
        for (int c = t; c < DMODEL; c += 128)
            acc += hr[c] * scale * w[c] * out_w[(size_t)c * 8 + j];
        red[t] = acc; __syncthreads();
        for (int s = 64; s > 0; s >>= 1) { if (t < s) red[t] += red[t + s]; __syncthreads(); }
        if (t == 0) out[(size_t)l * 8 + j] = red[0] + out_b[j];
        __syncthreads();
    }
}

// ---------------------------------------------------------------------------
// Host-side helpers
// ---------------------------------------------------------------------------
static inline void launch_gemm(const bf16_t* A, int lda, const bf16_t* B, int ldb,
                               float* C, int ldc, const float* bias,
                               int M, int N, int K, float alpha, bool transb,
                               hipStream_t st) {
    dim3 grid(N / BN, M / BM);
    if (transb)
        k_gemm_wmma<true><<<grid, 256, 0, st>>>(A, lda, B, ldb, C, ldc, bias, M, N, K, alpha);
    else
        k_gemm_wmma<false><<<grid, 256, 0, st>>>(A, lda, B, ldb, C, ldc, bias, M, N, K, alpha);
}

static inline void cvt(const float* in, bf16_t* out, int n, hipStream_t st) {
    k_f32_to_bf16<<<(n + 255) / 256, 256, 0, st>>>(in, out, n);
}

extern "C" void kernel_launch(void* const* d_in, const int* in_sizes, int n_in,
                              void* d_out, int out_size, void* d_ws, size_t ws_size,
                              hipStream_t stream) {
    (void)in_sizes; (void)n_in; (void)out_size; (void)ws_size;

    // ---- inputs (flattened dict order) ----
    const float* x          = (const float*)d_in[0];
    const int*   seperator  = (const int*)  d_in[1];
    const int*   tickers    = (const int*)  d_in[2];
    const float* freqs_cos  = (const float*)d_in[3];
    const float* freqs_sin  = (const float*)d_in[4];
    const float* sep_emb    = (const float*)d_in[5];
    const float* ticker_emb = (const float*)d_in[6];
    const float* shared_w   = (const float*)d_in[7];
    const float* shared_b   = (const float*)d_in[8];
    const float* unique_w   = (const float*)d_in[9];
    const float* unique_b   = (const float*)d_in[10];
    const float* final_norm = (const float*)d_in[11];
    const float* out_w      = (const float*)d_in[12];
    const float* out_b      = (const float*)d_in[13];
    auto LP = [&](int layer, int idx) -> const float* {
        return (const float*)d_in[14 + layer * 19 + idx];
    };
    // idx: 0 norm1, 1 norm2, 2 kvd_w, 3 kvd_b, 4 qd_w, 5 qd_b, 6 kvnorm, 7 qnorm,
    //      8 kvup_w, 9 kvup_b, 10 qup_w, 11 qup_b, 12 attn_norm, 13 o_w, 14 o_b,
    //      15 ffin_w, 16 ffin_b, 17 ffout_w, 18 ffout_b

    // ---- workspace carve-out ----
    char* ws = (char*)d_ws;
    size_t off = 0;
    auto alloc = [&](size_t bytes) -> char* {
        char* p = ws + off;
        off += (bytes + 255) & ~(size_t)255;
        return p;
    };
    bf16_t *w_kvd[2], *w_qd[2], *w_kvup[2], *w_qup[2], *w_o[2], *w_ffin[2], *w_ffout[2];
    for (int ly = 0; ly < 2; ++ly) {
        w_kvd[ly]   = (bf16_t*)alloc((size_t)DMODEL * 576 * 2);
        w_qd[ly]    = (bf16_t*)alloc((size_t)DMODEL * QC * 2);
        w_kvup[ly]  = (bf16_t*)alloc((size_t)KVC * 4096 * 2);
        w_qup[ly]   = (bf16_t*)alloc((size_t)QC * 3072 * 2);
        w_o[ly]     = (bf16_t*)alloc((size_t)2048 * DMODEL * 2);
        w_ffin[ly]  = (bf16_t*)alloc((size_t)DMODEL * 2 * DFF * 2);
        w_ffout[ly] = (bf16_t*)alloc((size_t)DFF * DMODEL * 2);
    }
    float*  h      = (float*) alloc((size_t)LTOT * DMODEL * 4);
    bf16_t* xn     = (bf16_t*)alloc((size_t)LTOT * DMODEL * 2);
    float*  ckv    = (float*) alloc((size_t)LTOT * 576 * 4);
    bf16_t* ckvn   = (bf16_t*)alloc((size_t)LTOT * KVC * 2);
    float*  kvb    = (float*) alloc((size_t)LTOT * 4096 * 4);
    float*  qdout  = (float*) alloc((size_t)LTOT * QC * 4);
    bf16_t* qn     = (bf16_t*)alloc((size_t)LTOT * QC * 2);
    float*  qf     = (float*) alloc((size_t)LTOT * 3072 * 4);
    bf16_t* Kattn  = (bf16_t*)alloc((size_t)NHEAD * LTOT * 192 * 2);
    bf16_t* Qattn  = (bf16_t*)alloc((size_t)NHEAD * LTOT * 192 * 2);
    bf16_t* Vattn  = (bf16_t*)alloc((size_t)NHEAD * LTOT * HD * 2);
    float*  scores = (float*) alloc((size_t)LTOT * LTOT * 4);
    bf16_t* probs  = (bf16_t*)alloc((size_t)LTOT * LTOT * 2);
    float*  octx   = (float*) alloc((size_t)LTOT * 2048 * 4);
    bf16_t* onorm  = (bf16_t*)alloc((size_t)LTOT * 2048 * 2);
    float*  gbuf   = (float*) alloc((size_t)LTOT * 2 * DFF * 4);
    bf16_t* gated  = (bf16_t*)alloc((size_t)LTOT * DFF * 2);
    float*  tmp    = (float*) alloc((size_t)LTOT * DMODEL * 4);

    // ---- convert weights f32 -> bf16 ----
    for (int ly = 0; ly < 2; ++ly) {
        cvt(LP(ly, 2),  w_kvd[ly],   DMODEL * 576,    stream);
        cvt(LP(ly, 4),  w_qd[ly],    DMODEL * QC,     stream);
        cvt(LP(ly, 8),  w_kvup[ly],  KVC * 4096,      stream);
        cvt(LP(ly, 10), w_qup[ly],   QC * 3072,       stream);
        cvt(LP(ly, 13), w_o[ly],     2048 * DMODEL,   stream);
        cvt(LP(ly, 15), w_ffin[ly],  DMODEL * 2 * DFF, stream);
        cvt(LP(ly, 17), w_ffout[ly], DFF * DMODEL,    stream);
    }

    // ---- embedding ----
    k_embed<<<(LTOT * DMODEL + 255) / 256, 256, 0, stream>>>(
        x, seperator, tickers, shared_w, shared_b, unique_w, unique_b,
        sep_emb, ticker_emb, h);

    // ---- transformer layers ----
    for (int ly = 0; ly < 2; ++ly) {
        k_rmsnorm_bf16<<<LTOT, 128, 0, stream>>>(h, DMODEL, xn, LP(ly, 0), DMODEL, EPS_DEF);

        // ckv = xn @ kvd_w + b     (2048 x 576)
        launch_gemm(xn, DMODEL, w_kvd[ly], 576, ckv, 576, LP(ly, 3),
                    LTOT, 576, DMODEL, 1.0f, false, stream);
        k_rmsnorm_bf16<<<LTOT, 128, 0, stream>>>(ckv, 576, ckvn, LP(ly, 6), KVC, EPS_DEF);
        // kv = ckvn @ kvup_w + b   (2048 x 4096)
        launch_gemm(ckvn, KVC, w_kvup[ly], 4096, kvb, 4096, LP(ly, 9),
                    LTOT, 4096, KVC, 1.0f, false, stream);

        // qd -> norm -> qup
        launch_gemm(xn, DMODEL, w_qd[ly], QC, qdout, QC, LP(ly, 5),
                    LTOT, QC, DMODEL, 1.0f, false, stream);
        k_rmsnorm_bf16<<<LTOT, 128, 0, stream>>>(qdout, QC, qn, LP(ly, 7), QC, EPS_DEF);
        launch_gemm(qn, QC, w_qup[ly], 3072, qf, 3072, LP(ly, 11),
                    LTOT, 3072, QC, 1.0f, false, stream);

        // build roped K/Q and V (bf16 attention operands)
        k_build_k<<<(NHEAD * LTOT * 192 + 255) / 256, 256, 0, stream>>>(
            kvb, ckv, freqs_cos, freqs_sin, Kattn);
        k_build_v<<<(NHEAD * LTOT * HD + 255) / 256, 256, 0, stream>>>(kvb, Vattn);
        k_build_q<<<(NHEAD * LTOT * 192 + 255) / 256, 256, 0, stream>>>(
            qf, freqs_cos, freqs_sin, Qattn);

        // attention, one head at a time (score buffer reused)
        for (int hh = 0; hh < NHEAD; ++hh) {
            const bf16_t* Qh = Qattn + (size_t)hh * LTOT * 192;
            const bf16_t* Kh = Kattn + (size_t)hh * LTOT * 192;
            const bf16_t* Vh = Vattn + (size_t)hh * LTOT * HD;
            // S = Q . K^T   (2048 x 2048, K=192)
            launch_gemm(Qh, 192, Kh, 192, scores, LTOT, nullptr,
                        LTOT, LTOT, 192, 1.0f, true, stream);
            // P = softmax(mask(S * scale))
            k_softmax<<<LTOT, 256, 0, stream>>>(scores, probs, ATTN_SCALE);
            // O_h = P . V   -> octx[:, hh*128 : hh*128+128]
            launch_gemm(probs, LTOT, Vh, HD, octx + (size_t)hh * HD, 2048, nullptr,
                        LTOT, HD, LTOT, 1.0f, false, stream);
        }

        // per-head rmsnorm over HD, then output proj + residual
        k_rmsnorm_bf16<<<LTOT * NHEAD, 128, 0, stream>>>(octx, HD, onorm, LP(ly, 12),
                                                         HD, EPS_ATTN);
        launch_gemm(onorm, 2048, w_o[ly], DMODEL, tmp, DMODEL, LP(ly, 14),
                    LTOT, DMODEL, 2048, 1.0f, false, stream);
        k_add<<<(LTOT * DMODEL + 255) / 256, 256, 0, stream>>>(h, tmp, LTOT * DMODEL);

        // FFN (SwiGLU)
        k_rmsnorm_bf16<<<LTOT, 128, 0, stream>>>(h, DMODEL, xn, LP(ly, 1), DMODEL, EPS_DEF);
        launch_gemm(xn, DMODEL, w_ffin[ly], 2 * DFF, gbuf, 2 * DFF, LP(ly, 16),
                    LTOT, 2 * DFF, DMODEL, 1.0f, false, stream);
        k_swiglu<<<(LTOT * DFF + 255) / 256, 256, 0, stream>>>(gbuf, gated);
        launch_gemm(gated, DFF, w_ffout[ly], DMODEL, tmp, DMODEL, LP(ly, 18),
                    LTOT, DMODEL, DFF, 1.0f, false, stream);
        k_add<<<(LTOT * DMODEL + 255) / 256, 256, 0, stream>>>(h, tmp, LTOT * DMODEL);
    }

    // ---- final norm + output head ----
    k_final_out<<<LTOT, 128, 0, stream>>>(h, final_norm, out_w, out_b, (float*)d_out);
}
